// TestAllReduceFusedAddRMSNormStaticQuantModel_41953240547417
// MI455X (gfx1250) — 8-bit matrix operand order repaired and checked by reading, compiled, not run on hardware
//
#include <hip/hip_runtime.h>
#include <hip/hip_bf16.h>
#include <stdint.h>

#define T_DIM 8192
#define H_DIM 4096
#define I_DIM 8192
#define FP8_MAXF 448.0f

#define BM 128
#define BN 256
#define BK 128
#define NK (H_DIM / BK)

typedef int   v16i __attribute__((ext_vector_type(16)));
typedef float v8f  __attribute__((ext_vector_type(8)));

#if __has_builtin(__builtin_amdgcn_cvt_pk_fp8_f32)
#define USE_HW_FP8_CVT 1
#else
#define USE_HW_FP8_CVT 0
#endif

__device__ __forceinline__ float clamp448(float x) {
    return fminf(fmaxf(x, -FP8_MAXF), FP8_MAXF);
}

__device__ __forceinline__ unsigned char sw_f32_to_e4m3(float x) {
    unsigned int u = __float_as_uint(x);
    unsigned char s = (unsigned char)((u >> 24) & 0x80u);
    float a = fabsf(x);
    a = fminf(a, FP8_MAXF);
    if (a < 0.0009765625f)
        return s;
    int e;
    (void)frexpf(a, &e);
    int E = e - 1;
    if (E < -6) E = -6;
    float step = exp2f((float)(E - 3));
    float r = rintf(a / step);
    if (r >= 16.0f) { r *= 0.5f; E += 1; }
    unsigned char bits;
    if (r < 8.0f) bits = (unsigned char)r;
    else          bits = (unsigned char)(((E + 7) << 3) | ((int)r - 8));
    return (unsigned char)(s | bits);
}

__device__ __forceinline__ unsigned int pack4_e4m3(float x0, float x1, float x2, float x3) {
#if USE_HW_FP8_CVT
    int p = __builtin_amdgcn_cvt_pk_fp8_f32(x0, x1, 0, false);
    p     = __builtin_amdgcn_cvt_pk_fp8_f32(x2, x3, p, true);
    return (unsigned int)p;
#else
    return (unsigned int)sw_f32_to_e4m3(x0)
         | ((unsigned int)sw_f32_to_e4m3(x1) << 8)
         | ((unsigned int)sw_f32_to_e4m3(x2) << 16)
         | ((unsigned int)sw_f32_to_e4m3(x3) << 24);
#endif
}

__device__ __forceinline__ void async_copy_b128(const unsigned char* g, unsigned char* l) {
    unsigned int loff = (unsigned int)(unsigned long long)(uintptr_t)l;
    unsigned long long gaddr = (unsigned long long)(uintptr_t)g;
    asm volatile("global_load_async_to_lds_b128 %0, %1, off"
                 :: "v"(loff), "v"(gaddr) : "memory");
}
#define WAIT_ASYNCCNT_12() asm volatile("s_wait_asynccnt 0xc" ::: "memory")
#define WAIT_ASYNCCNT_0()  asm volatile("s_wait_asynccnt 0x0" ::: "memory")

__global__ __launch_bounds__(256)
void quant_w_transpose_kernel(const float* __restrict__ w, unsigned char* __restrict__ wqT) {
    __shared__ unsigned char tile[64][68];
    const int t  = threadIdx.x;
    const int k0 = blockIdx.y * 64;
    const int n0 = blockIdx.x * 64;
    const int cn = (t & 15) * 4;
    const int rk = t >> 4;

#pragma unroll
    for (int p = 0; p < 4; ++p) {
        const int row = rk + p * 16;
        const float4 f = *(const float4*)(w + (size_t)(k0 + row) * I_DIM + n0 + cn);
        *(unsigned int*)&tile[row][cn] = pack4_e4m3(f.x, f.y, f.z, f.w);
    }
    __syncthreads();

    const int ck = (t & 15) * 4;
    const int rn = t >> 4;
#pragma unroll
    for (int p = 0; p < 4; ++p) {
        const int nrow = rn + p * 16;
        unsigned int v = (unsigned int)tile[ck + 0][nrow]
                       | ((unsigned int)tile[ck + 1][nrow] << 8)
                       | ((unsigned int)tile[ck + 2][nrow] << 16)
                       | ((unsigned int)tile[ck + 3][nrow] << 24);
        *(unsigned int*)(wqT + (size_t)(n0 + nrow) * H_DIM + k0 + ck) = v;
    }
}

__global__ __launch_bounds__(256)
void rmsnorm_quant_kernel(const float* __restrict__ hs, const float* __restrict__ res,
                          const float* __restrict__ nw, const float* __restrict__ scale,
                          float* __restrict__ residual_out, unsigned char* __restrict__ q) {
    __shared__ float red[256];
    const int t = threadIdx.x;
    const size_t base = (size_t)blockIdx.x * H_DIM;
    const int c0 = t * 16;

    float4 v[4];
    float ss = 0.0f;
#pragma unroll
    for (int p = 0; p < 4; ++p) {
        float4 h = *(const float4*)(hs + base + c0 + p * 4);
        float4 r = *(const float4*)(res + base + c0 + p * 4);
        float4 x; x.x = h.x + r.x; x.y = h.y + r.y; x.z = h.z + r.z; x.w = h.w + r.w;
        v[p] = x;
        ss += x.x * x.x + x.y * x.y + x.z * x.z + x.w * x.w;
    }
    red[t] = ss;
#pragma unroll
    for (int sft = 128; sft > 0; sft >>= 1) {
        __syncthreads();
        if (t < sft) red[t] += red[t + sft];
    }
    __syncthreads();
    const float inv  = rsqrtf(red[0] * (1.0f / (float)H_DIM) + 1e-6f);
    const float invs = 1.0f / scale[0];

    unsigned int qw[4];
#pragma unroll
    for (int p = 0; p < 4; ++p) {
        *(float4*)(residual_out + base + c0 + p * 4) = v[p];
        float4 g = *(const float4*)(nw + c0 + p * 4);
        float n0 = clamp448(v[p].x * inv * g.x * invs);
        float n1 = clamp448(v[p].y * inv * g.y * invs);
        float n2 = clamp448(v[p].z * inv * g.z * invs);
        float n3 = clamp448(v[p].w * inv * g.w * invs);
        qw[p] = pack4_e4m3(n0, n1, n2, n3);
    }
    uint4 packed; packed.x = qw[0]; packed.y = qw[1]; packed.z = qw[2]; packed.w = qw[3];
    *(uint4*)(q + base + c0) = packed;
}

union Frag {
    v16i  v;
    uint2 q2[8];
    uint4 q4[4];
};

__global__ __launch_bounds__(256)
void gemm_fp8_kernel(const unsigned char* __restrict__ qa,
                     const unsigned char* __restrict__ wqT,
                     const float* __restrict__ scale, const float* __restrict__ wscale,
                     float* __restrict__ out) {
    __shared__ unsigned char As[2][BM * BK];
    __shared__ unsigned char Bs[2][BN * BK];

    const int t     = threadIdx.x;
    const int lane  = t & 31;
    const int wave  = t >> 5;
    const int waveM = wave & 1;
    const int waveN = wave >> 1;
    const int half  = (lane >> 4) & 1;
    const int l16   = lane & 15;

    const size_t m0 = (size_t)blockIdx.y * BM;
    const size_t n0 = (size_t)blockIdx.x * BN;

    v8f acc[4][4];
#pragma unroll
    for (int i = 0; i < 4; ++i)
#pragma unroll
        for (int j = 0; j < 4; ++j) {
            v8f z = {0.f, 0.f, 0.f, 0.f, 0.f, 0.f, 0.f, 0.f};
            acc[i][j] = z;
        }

    const int sRow = t >> 3;
    const int sCol = (t & 7) * 16;
    const unsigned char* gA = qa  + (m0 + sRow) * H_DIM + sCol;
    const unsigned char* gB = wqT + (n0 + sRow) * H_DIM + sCol;

    auto stage = [&](int buf, int k0) {
#pragma unroll
        for (int r = 0; r < 4; ++r)
            async_copy_b128(gA + k0 + (size_t)r * 32 * H_DIM,
                            &As[buf][(sRow + r * 32) * BK + sCol]);
#pragma unroll
        for (int r = 0; r < 8; ++r)
            async_copy_b128(gB + k0 + (size_t)r * 32 * H_DIM,
                            &Bs[buf][(sRow + r * 32) * BK + sCol]);
    };

    stage(0, 0);

    for (int kt = 0; kt < NK; ++kt) {
        const int cur = kt & 1;
        if (kt + 1 < NK) {
            stage(cur ^ 1, (kt + 1) * BK);
            WAIT_ASYNCCNT_12();
        } else {
            WAIT_ASYNCCNT_0();
        }
        __syncthreads();

        Frag a[4];
#pragma unroll
        for (int i = 0; i < 4; ++i) {
            const unsigned char* ap = &As[cur][(waveM * 64 + i * 16 + l16) * BK];
#pragma unroll
            for (int j = 0; j < 8; ++j)
                a[i].q2[j] = *(const uint2*)(ap + (j >> 1) * 32 + (j & 1) * 8 + half * 16);
        }
#pragma unroll
        for (int j = 0; j < 4; ++j) {
            Frag b;
            const unsigned char* bp = &Bs[cur][(waveN * 64 + j * 16 + l16) * BK];
#pragma unroll
            for (int g = 0; g < 4; ++g)
                b.q4[g] = *(const uint4*)(bp + g * 32 + half * 16);
#pragma unroll
            for (int i = 0; i < 4; ++i)
                acc[i][j] = __builtin_amdgcn_wmma_f32_16x16x128_fp8_fp8(
                    a[i].v, b.v, (short)0, acc[i][j], false, false);
        }

        __syncthreads();
    }

    const float s = scale[0] * wscale[0];
#pragma unroll
    for (int i = 0; i < 4; ++i)
#pragma unroll
        for (int j = 0; j < 4; ++j) {
            const size_t gn = n0 + waveN * 64 + j * 16 + l16;
#pragma unroll
            for (int v = 0; v < 8; ++v) {
                const size_t gm = m0 + waveM * 64 + i * 16 + half * 8 + v;
                out[gm * I_DIM + gn] = acc[i][j][v] * s;
            }
        }
}

extern "C" void kernel_launch(void* const* d_in, const int* in_sizes, int n_in,
                              void* d_out, int out_size, void* d_ws, size_t ws_size,
                              hipStream_t stream) {
    const float* hs     = (const float*)d_in[0];
    const float* res    = (const float*)d_in[1];
    const float* nw     = (const float*)d_in[2];
    const float* w      = (const float*)d_in[3];
    const float* scale  = (const float*)d_in[4];
    const float* wscale = (const float*)d_in[5];

    float* out          = (float*)d_out;
    float* residual_out = out + (size_t)T_DIM * I_DIM;

    unsigned char* qa  = (unsigned char*)d_ws;
    unsigned char* wqT = qa + (size_t)T_DIM * H_DIM;

    quant_w_transpose_kernel<<<dim3(I_DIM / 64, H_DIM / 64), 256, 0, stream>>>(w, wqT);
    rmsnorm_quant_kernel<<<T_DIM, 256, 0, stream>>>(hs, res, nw, scale, residual_out, qa);
    gemm_fp8_kernel<<<dim3(I_DIM / BN, T_DIM / BM), 256, 0, stream>>>(qa, wqT, scale, wscale, out);
}
